// SE3Transformer_29618094473365
// MI455X (gfx1250) — compile-verified, matrix operand order checked by reading, not run on hardware
//
// SE(3)-Transformer forward for MI455X (gfx1250, wave32).
// Dense math on v_wmma_f32_16x16x32_bf16 (bf16 in, f32 accumulate).
// WMMA B-operands pre-packed into fragment-ordered bf16 (2x global_load_b128
// per lane); A-side loads clamped, never predicated (packed-B zero padding
// absorbs OOB garbage). Edge-kernel gathers use GLOBAL_LOAD_ASYNC_TO_LDS
// (ASYNCcnt) when the toolchain exposes the builtins, overlapping the gather
// with per-edge geometry register work. Segment ops use global f32 atomics into
// L2-resident accumulators. Needs ~70 MB of d_ws scratch.
#include <hip/hip_runtime.h>
#include <math.h>
#include <stdint.h>

#define DEV __device__ __forceinline__

#if defined(__gfx1250__) && \
    __has_builtin(__builtin_amdgcn_global_load_async_to_lds_b128) && \
    __has_builtin(__builtin_amdgcn_global_load_async_to_lds_b32) && \
    __has_builtin(__builtin_amdgcn_s_wait_asynccnt)
#define HAVE_ASYNC_LDS 1
#else
#define HAVE_ASYNC_LDS 0
#endif

typedef __attribute__((ext_vector_type(16))) __bf16   v16bf;
typedef __attribute__((ext_vector_type(2)))  __bf16   bf16x2;
typedef __attribute__((ext_vector_type(8)))  float    v8f;
typedef __attribute__((ext_vector_type(8)))  uint32_t v8u;
typedef __attribute__((ext_vector_type(4)))  float    f4;
typedef __attribute__((ext_vector_type(2)))  float    f2;

// K position held by VGPR-half v for a lane in K-half kh (16-bit A/B fragment layout)
DEV int akb(int v, int kh) { return (v < 4) ? (kh * 8 + 2 * v) : (16 + kh * 8 + 2 * (v - 4)); }

DEV float warp_sum(float v) {
#pragma unroll
  for (int m = 16; m > 0; m >>= 1) v += __shfl_xor(v, m, 32);
  return v;
}

DEV void atomAddF(float* p, float v) { unsafeAtomicAdd(p, v); }

// Wave-local LDS ordering: LDS ops are in-order per wave (DScnt); this just
// stops the compiler from reordering across the producer/consumer point.
DEV void wave_fence() {
  __builtin_amdgcn_wave_barrier();
  __threadfence_block();
  __builtin_amdgcn_wave_barrier();
}

#if HAVE_ASYNC_LDS
// Builtin signatures (from clang diagnostics): param0 is a NON-const typed
// pointer in AS(1): b128 -> int __attribute__((vector_size(16)))*, b32 -> int*.
typedef int v4i __attribute__((vector_size(16)));
typedef __attribute__((address_space(1))) v4i  gv4i;
typedef __attribute__((address_space(3))) v4i  lv4i;
typedef __attribute__((address_space(1))) int  gi32;
typedef __attribute__((address_space(3))) int  li32;
// Per-lane async copy: LDS[dst] = MEM[src] (b128/b32), tracked by ASYNCcnt.
DEV void async_ld128(const float* g, float* l) {
  __builtin_amdgcn_global_load_async_to_lds_b128((gv4i*)g, (lv4i*)l, 0, 0);
}
DEV void async_ld32(const float* g, float* l) {
  __builtin_amdgcn_global_load_async_to_lds_b32((gi32*)g, (li32*)l, 0, 0);
}
#endif

// ---------------------------------------------------------------------------
// packw: convert a strided f32 (K x N) matrix into WMMA-fragment-ordered bf16.
// Layout: chunk (kc, tn) -> 32 lanes x 8 dwords (two b128 per lane at use).
// OOB K/N entries are packed as 0 (this is what makes A-side clamping safe).
// ---------------------------------------------------------------------------
__global__ __launch_bounds__(128) void packw_kernel(const float* B, long lbr, long lbc,
                                                    int K, int N, uint32_t* out) {
  const int wid  = (blockIdx.x * blockDim.x + threadIdx.x) >> 5;
  const int lane = threadIdx.x & 31;
  const int tilesN = (N + 15) >> 4, kchunks = (K + 31) >> 5;
  if (wid >= tilesN * kchunks) return;
  const int kc = wid / tilesN, tn = wid - kc * tilesN;
  const int col = tn * 16 + (lane & 15), kh = lane >> 4;
  const int cc = col < N ? col : 0;
  v8u o;
#pragma unroll
  for (int v = 0; v < 8; ++v) {
    int k1 = kc * 32 + akb(v, kh), k2 = k1 + 1;
    int k1c = k1 < K ? k1 : 0, k2c = k2 < K ? k2 : 0;
    float b0 = B[(size_t)k1c * lbr + (size_t)cc * lbc];
    float b1 = B[(size_t)k2c * lbr + (size_t)cc * lbc];
    b0 = (k1 < K && col < N) ? b0 : 0.f;
    b1 = (k2 < K && col < N) ? b1 : 0.f;
    bf16x2 p; p[0] = (__bf16)b0; p[1] = (__bf16)b1;
    o[v] = __builtin_bit_cast(uint32_t, p);
  }
  ((v8u*)out)[(size_t)wid * 32 + lane] = o;
}

// packr: per-lane radial-MLP first layer {b1, W1 rows} in fragment-K order,
// so the edge kernel builds its A fragment from contiguous float4 loads.
__global__ void packr_kernel(const float* W1, const float* b1, float* out) {
  int i = blockIdx.x * blockDim.x + threadIdx.x;
  if (i >= 512) return;
  int lane = i >> 4, slot = i & 15;
  int v = slot >> 1, s = slot & 1, kh = lane >> 4;
  int k = akb(v, kh) + s;
  f4 o; o[0] = b1[k]; o[1] = W1[k]; o[2] = W1[32 + k]; o[3] = W1[64 + k];
  ((f4*)out)[i] = o;
}

// ---------------------------------------------------------------------------
// Generic strided GEMM: C = act(A[|A2 concat] @ B + bias) (+ Cadd)
// One wave per 16x16 tile; B pre-packed bf16 (2x b128 per lane per K-chunk);
// A loads clamped, never predicated (packed-B zeros absorb OOB garbage).
// ---------------------------------------------------------------------------
struct GemmArgs {
  const float* A;  long lar, lac;
  const float* A2; long la2r, la2c; int K1;   // A2 supplies k >= K1 (concat)
  const uint32_t* Bpk;
  const float* bias;
  const float* Cadd; long lcar, lcac;
  float* C; long lcr, lcc;
  int M, N, K, act;                            // act: 0 none, 1 relu, 2 elu
};

DEV float ldA(const GemmArgs& g, int m, int k) {
  bool useA2 = g.A2 && (k >= g.K1);
  int kk = useA2 ? k - g.K1 : k;
  int kmax = useA2 ? (g.K - g.K1) : (g.A2 ? g.K1 : g.K);
  kk = kk < kmax ? kk : kmax - 1;
  const float* p = useA2 ? g.A2 : g.A;
  long lr = useA2 ? g.la2r : g.lar;
  long lc = useA2 ? g.la2c : g.lac;
  int mc = m < g.M ? m : g.M - 1;
  return p[(size_t)mc * lr + (size_t)kk * lc];
}

__global__ __launch_bounds__(128) void gemm_kernel(GemmArgs g) {
  const int wid  = (blockIdx.x * blockDim.x + threadIdx.x) >> 5;
  const int lane = threadIdx.x & 31;
  const int tilesN = (g.N + 15) >> 4;
  const int tm = (wid / tilesN) * 16, tn16 = wid % tilesN;
  if (tm >= g.M) return;                       // whole-wave uniform exit
  const int row = lane & 15, kh = lane >> 4;
  const int mrow = (tm + row < g.M) ? tm + row : g.M - 1;   // clamped only
  const float* Arow = g.A + (size_t)mrow * g.lar;
  const bool fastA = (g.lac == 1) && !g.A2 && ((g.lar & 1) == 0);
  const int kchunks = (g.K + 31) >> 5;
  v8f acc = {};
  for (int kc = 0; kc < kchunks; ++kc) {
    const int k0 = kc * 32;
    if (kc + 1 < kchunks)
      __builtin_prefetch(g.Bpk + ((size_t)(kc + 1) * tilesN + tn16) * 256, 0, 0);
    v16bf a;
    if (fastA && (k0 + 32 <= g.K)) {
#pragma unroll
      for (int v = 0; v < 8; ++v) {
        f2 x = *(const f2*)(Arow + k0 + akb(v, kh));
        a[2 * v] = (__bf16)x[0]; a[2 * v + 1] = (__bf16)x[1];
      }
    } else {
#pragma unroll
      for (int v = 0; v < 8; ++v) {
        int kk1 = k0 + akb(v, kh);
        a[2 * v]     = (__bf16)ldA(g, tm + row, kk1);
        a[2 * v + 1] = (__bf16)ldA(g, tm + row, kk1 + 1);
      }
    }
    v8u bw = ((const v8u*)g.Bpk)[((size_t)kc * tilesN + tn16) * 32 + lane];
    v16bf b = __builtin_bit_cast(v16bf, bw);
    acc = __builtin_amdgcn_wmma_f32_16x16x32_bf16(false, a, false, b, (short)0, acc, false, false);
  }
  const int col = tn16 * 16 + row, mb = kh * 8;
  if (col < g.N) {
    float bv = g.bias ? g.bias[col] : 0.f;
#pragma unroll
    for (int r = 0; r < 8; ++r) {
      int m = tm + mb + r;
      if (m >= g.M) break;
      float x = acc[r] + bv;
      if (g.act == 1) x = fmaxf(x, 0.f);
      else if (g.act == 2 && x < 0.f) x = __expf(x) - 1.f;
      if (g.Cadd) x += g.Cadd[(size_t)m * g.lcar + (size_t)col * g.lcac];
      g.C[(size_t)m * g.lcr + (size_t)col * g.lcc] = x;
    }
  }
}

// ---------------------------------------------------------------------------
// Fused per-edge engine: radial hidden (VALU from packed W1), hidden@W2 via
// WMMA with pre-packed bf16 B fragments, equivariant basis contraction fused
// per tile ((j,o,c) decomposed with shifts: cin/cout are powers of two).
//   MODE 0: attention logits    MODE 1: attn-weighted scatter   MODE 2: conv
// 16 edges per wave; gathers staged to LDS asynchronously (ASYNCcnt).
// ---------------------------------------------------------------------------
struct EdgePair {
  const float* b2; const uint32_t* W2pk; const float* R1pk;
  int cin, cout, cinLog, coutLog, nout, di, dof;
};
struct EdgeArgs {
  const float* rel; const float* w; const int* src; const int* dst; int E;
  const float* h0; const float* h1; int c1;
  EdgePair pr[4]; int np;
  const float* q0; const float* q1; float isdk; float* logits; int shared1;  // MODE 0
  const float* attn; float* agg0; float* agg1;                                // MODE 1
  float* acc0; float* acc1; float* deg;                                       // MODE 2
};

template <int MODE, int C0, int C1>
__global__ __launch_bounds__(128) void edge_kernel(EdgeArgs A) {
  constexpr int C1P = (C1 > 0 ? C1 + 1 : 1);
  constexpr int C1S = (C1 > 0 ? C1 : 1) * 3;   // clamped divisor (avoids /0 warn)
  __shared__ __align__(16) float sf0[4][16][32];
  __shared__ __align__(16) float sf1[4][16][32][3];
  __shared__ float sRt[4][16][17];
  __shared__ float sO0[4][16][C0 + 1];
  __shared__ float sO1[4][16][C1P][3];
  const int wv = threadIdx.x >> 5, lane = threadIdx.x & 31;
  const int erow = lane & 15, kh = lane >> 4;
  const int e0 = (blockIdx.x * 4 + wv) * 16;
  if (e0 >= A.E) return;                        // waves are fully independent
  const int eg = (e0 + erow < A.E) ? (e0 + erow) : (A.E - 1);
  const bool ev = (e0 + erow) < A.E;

  // Kick off the source-feature gather first so it overlaps the register work
  // below (async global->LDS when available, tracked by ASYNCcnt).
#if HAVE_ASYNC_LDS
  for (int i = lane; i < 16 * 8; i += 32) {
    int el = i >> 3, c4 = i & 7;
    int ee = (e0 + el < A.E) ? (e0 + el) : (A.E - 1);
    async_ld128(A.h0 + (size_t)A.src[ee] * 32 + c4 * 4, &sf0[wv][el][c4 * 4]);
  }
  if (A.c1 == 32) {
    for (int i = lane; i < 16 * 24; i += 32) {
      int el = i / 24, c4 = i - el * 24;
      int ee = (e0 + el < A.E) ? (e0 + el) : (A.E - 1);
      async_ld128(A.h1 + (size_t)A.src[ee] * 96 + c4 * 4, (float*)sf1[wv][el] + c4 * 4);
    }
  } else if (A.c1 == 1) {
    for (int i = lane; i < 48; i += 32) {
      int el = i / 3, m = i - el * 3;
      int ee = (e0 + el < A.E) ? (e0 + el) : (A.E - 1);
      async_ld32(A.h1 + (size_t)A.src[ee] * 3 + m, &sf1[wv][el][0][m]);
    }
  }
#else
  for (int i = lane; i < 16 * 8; i += 32) {
    int el = i >> 3, c4 = i & 7;
    int ee = (e0 + el < A.E) ? (e0 + el) : (A.E - 1);
    ((f4*)sf0[wv][el])[c4] = ((const f4*)(A.h0 + (size_t)A.src[ee] * 32))[c4];
  }
  if (A.c1 == 32) {
    for (int i = lane; i < 16 * 24; i += 32) {
      int el = i / 24, c4 = i - el * 24;
      int ee = (e0 + el < A.E) ? (e0 + el) : (A.E - 1);
      ((f4*)sf1[wv][el])[c4] = ((const f4*)(A.h1 + (size_t)A.src[ee] * 96))[c4];
    }
  } else if (A.c1 == 1) {
    for (int i = lane; i < 48; i += 32) {
      int el = i / 3, m = i - el * 3;
      int ee = (e0 + el < A.E) ? (e0 + el) : (A.E - 1);
      sf1[wv][el][0][m] = A.h1[(size_t)A.src[ee] * 3 + m];
    }
  }
#endif

  // per-edge geometry + radial features (w0, w1, |rel|); both halves hold
  // their own edge's u in registers, so no LDS needed for it.
  float rx = A.rel[(size_t)eg * 3 + 0], ry = A.rel[(size_t)eg * 3 + 1], rz = A.rel[(size_t)eg * 3 + 2];
  float rr = sqrtf(rx * rx + ry * ry + rz * rz);
  float ivr = 1.f / (rr + 1e-8f);
  float u0 = rx * ivr, u1 = ry * ivr, u2 = rz * ivr;
  float ft0 = A.w[(size_t)eg * 2 + 0], ft1 = A.w[(size_t)eg * 2 + 1], ft2 = rr;

  for (int i = lane; i < 16 * C0; i += 32) sO0[wv][i / C0][i % C0] = 0.f;
  if (C1 > 0)
    for (int i = lane; i < 16 * C1 * 3; i += 32) {
      int el = i / C1S; int r = i - el * C1S;
      sO1[wv][el][r / 3][r % 3] = 0.f;
    }
  wave_fence();
#if HAVE_ASYNC_LDS
  __builtin_amdgcn_s_wait_asynccnt(0);   // gathered rows now visible in LDS
#endif

  for (int p = 0; p < A.np; ++p) {
    const EdgePair P = A.pr[p];
    // A fragment: radial hidden = relu(feat @ W1 + b1) from packed coefficients
    const f4* rp = (const f4*)P.R1pk + (size_t)lane * 16;
    v16bf af;
#pragma unroll
    for (int v = 0; v < 8; ++v) {
      f4 ca = rp[2 * v], cb = rp[2 * v + 1];
      float h1v = fmaxf(ca[0] + ft0 * ca[1] + ft1 * ca[2] + ft2 * ca[3], 0.f);
      float h2v = fmaxf(cb[0] + ft0 * cb[1] + ft1 * cb[2] + ft2 * cb[3], 0.f);
      af[2 * v] = (__bf16)h1v; af[2 * v + 1] = (__bf16)h2v;
    }
    const int ntiles = P.nout >> 4;
    for (int t = 0; t < ntiles; ++t) {
      if (t + 1 < ntiles) __builtin_prefetch(P.W2pk + (size_t)(t + 1) * 256, 0, 0);
      v8u bw = ((const v8u*)P.W2pk)[(size_t)t * 32 + lane];
      v16bf bf = __builtin_bit_cast(v16bf, bw);
      v8f c = {};
      c = __builtin_amdgcn_wmma_f32_16x16x32_bf16(false, af, false, bf, (short)0, c, false, false);
#pragma unroll
      for (int r = 0; r < 8; ++r) sRt[wv][kh * 8 + r][erow] = c[r];
      wave_fence();

      // consume: both halves take 8 columns each; e = erow for every lane
      const int e = erow, n0 = t * 16;
      const f4* b2p = (const f4*)(P.b2 + n0 + kh * 8);
      f4 bA = b2p[0], bB = b2p[1];
      if (P.cinLog == 5) {                     // cin == 32: (j,o) constant per tile
        const int c0 = n0 & 31, oj = n0 >> 5;
        const int o = oj & (P.cout - 1), j = oj >> P.coutLog;
        float s0 = 0.f, s1x = 0.f, s1y = 0.f, s1z = 0.f;
#pragma unroll
        for (int i = 0; i < 8; ++i) {
          int nn = kh * 8 + i, cch = c0 + nn;
          float R = sRt[wv][e][nn] + (i < 4 ? bA[i] : bB[i - 4]);
          if (P.di == 0) {
            s0 += R * sf0[wv][e][cch];         // also feeds (0,1): x u_m after combine
          } else {
            float fx = sf1[wv][e][cch][0], fy = sf1[wv][e][cch][1], fz = sf1[wv][e][cch][2];
            if (P.dof == 0) s0 += R * (u0 * fx + u1 * fy + u2 * fz);
            else if (j == 0) { s1x += R * fx; s1y += R * fy; s1z += R * fz; }
            else if (j == 1) { s1x += R * (-u2 * fy + u1 * fz);
                               s1y += R * ( u2 * fx - u0 * fz);
                               s1z += R * (-u1 * fx + u0 * fy); }
            else { float d = u0 * fx + u1 * fy + u2 * fz;
                   s1x += R * (3.f * u0 * d - fx);
                   s1y += R * (3.f * u1 * d - fy);
                   s1z += R * (3.f * u2 * d - fz); }
          }
        }
        if (P.dof == 0) {
          s0 += __shfl_xor(s0, 16, 32);
          if (lane < 16) sO0[wv][e][o] += s0;
        } else if (P.di == 0) {
          s0 += __shfl_xor(s0, 16, 32);
          if (lane < 16) { sO1[wv][e][o][0] += s0 * u0; sO1[wv][e][o][1] += s0 * u1; sO1[wv][e][o][2] += s0 * u2; }
        } else {
          s1x += __shfl_xor(s1x, 16, 32);
          s1y += __shfl_xor(s1y, 16, 32);
          s1z += __shfl_xor(s1z, 16, 32);
          if (lane < 16) { sO1[wv][e][o][0] += s1x; sO1[wv][e][o][1] += s1y; sO1[wv][e][o][2] += s1z; }
        }
      } else {                                 // cin == 1 (cout == 16): o = nn, j = t
        const int j = t;
        float fx = sf1[wv][e][0][0], fy = sf1[wv][e][0][1], fz = sf1[wv][e][0][2];
#pragma unroll
        for (int i = 0; i < 8; ++i) {
          int nn = kh * 8 + i, o = nn;
          float R = sRt[wv][e][nn] + (i < 4 ? bA[i] : bB[i - 4]);
          if (P.dof == 0) sO0[wv][e][o] += R * (u0 * fx + u1 * fy + u2 * fz);
          else if (j == 0) { sO1[wv][e][o][0] += R * fx; sO1[wv][e][o][1] += R * fy; sO1[wv][e][o][2] += R * fz; }
          else if (j == 1) { sO1[wv][e][o][0] += R * (-u2 * fy + u1 * fz);
                             sO1[wv][e][o][1] += R * ( u2 * fx - u0 * fz);
                             sO1[wv][e][o][2] += R * (-u1 * fx + u0 * fy); }
          else { float d = u0 * fx + u1 * fy + u2 * fz;
                 sO1[wv][e][o][0] += R * (3.f * u0 * d - fx);
                 sO1[wv][e][o][1] += R * (3.f * u1 * d - fy);
                 sO1[wv][e][o][2] += R * (3.f * u2 * d - fz); }
        }
      }
    }
  }
  wave_fence();

  if (lane < 16 && ev) {
    const int e = lane, dn = A.dst[eg];
    if (MODE == 0) {
      float l0 = 0.f, l1 = 0.f;
      for (int o = 0; o < 16; ++o) {
        float s = sO0[wv][e][o] * A.q0[(size_t)dn * 16 + o];
        if (o < 8) l0 += s; else l1 += s;
      }
      if (A.shared1)
        for (int o = 0; o < 16; ++o)
          for (int m = 0; m < 3; ++m) {
            float s = sO1[wv][e][o][m] * A.q1[((size_t)dn * 16 + o) * 3 + m];
            if (o < 8) l0 += s; else l1 += s;
          }
      A.logits[(size_t)eg * 2 + 0] = l0 * A.isdk;
      A.logits[(size_t)eg * 2 + 1] = l1 * A.isdk;
    } else if (MODE == 1) {
      float a0 = A.attn[(size_t)eg * 2 + 0], a1 = A.attn[(size_t)eg * 2 + 1];
      for (int o = 0; o < 16; ++o) {
        float a = (o < 8) ? a0 : a1;
        atomAddF(&A.agg0[(size_t)dn * 16 + o], a * sO0[wv][e][o]);
        for (int m = 0; m < 3; ++m)
          atomAddF(&A.agg1[((size_t)dn * 16 + o) * 3 + m], a * sO1[wv][e][o][m]);
      }
    } else {
      for (int o = 0; o < C0; ++o) atomAddF(&A.acc0[(size_t)dn * C0 + o], sO0[wv][e][o]);
      if (C1 > 0)
        for (int o = 0; o < C1; ++o)
          for (int m = 0; m < 3; ++m)
            atomAddF(&A.acc1[((size_t)dn * C1 + o) * 3 + m], sO1[wv][e][o][m]);
      atomAddF(&A.deg[dn], 1.f);
    }
  }
}

// ---------------------------------------------------------------------------
// Small elementwise / segment kernels
// ---------------------------------------------------------------------------
__global__ void fill_kernel(float* p, float v, long n) {
  long i = (long)blockIdx.x * blockDim.x + threadIdx.x;
  if (i < n) p[i] = v;
}

DEV unsigned enc_f(float f) { int b = __float_as_int(f); return (b >= 0) ? ((unsigned)b | 0x80000000u) : (unsigned)(~b); }
DEV float dec_f(unsigned u) { int b = (u & 0x80000000u) ? (int)(u & 0x7fffffffu) : ~(int)u; return __int_as_float(b); }

__global__ void segmax_kernel(const float* logits, const int* dst, unsigned* mx, long EH) {
  long i = (long)blockIdx.x * blockDim.x + threadIdx.x;
  if (i >= EH) return;
  int e = (int)(i >> 1), h = (int)(i & 1);
  atomicMax(&mx[(size_t)dst[e] * 2 + h], enc_f(logits[i]));
}
__global__ void segexp_kernel(const float* logits, const int* dst, const unsigned* mx,
                              float* ex, float* den, long EH) {
  long i = (long)blockIdx.x * blockDim.x + threadIdx.x;
  if (i >= EH) return;
  int e = (int)(i >> 1), h = (int)(i & 1), dn = dst[e];
  float x = __expf(logits[i] - dec_f(mx[(size_t)dn * 2 + h]));
  ex[i] = x;
  atomAddF(&den[(size_t)dn * 2 + h], x);
}
__global__ void attnnorm_kernel(const float* ex, const int* dst, const float* den, float* attn, long EH) {
  long i = (long)blockIdx.x * blockDim.x + threadIdx.x;
  if (i >= EH) return;
  int e = (int)(i >> 1), h = (int)(i & 1);
  attn[i] = ex[i] / (den[(size_t)dst[e] * 2 + h] + 1e-9f);
}

// Equivariant group norm: one wave per node, lane = channel (wave32 == 32 ch)
__global__ __launch_bounds__(128) void gnorm_kernel(const float* f, float* out,
                                                    const float* g, const float* b, int N, int M) {
  int wid = (blockIdx.x * blockDim.x + threadIdx.x) >> 5;
  int lane = threadIdx.x & 31;
  if (wid >= N) return;
  size_t base = ((size_t)wid * 32 + lane) * M;
  float nn = 0.f;
  for (int m = 0; m < M; ++m) { float v = f[base + m]; nn += v * v; }
  nn = sqrtf(nn);
  float mu = warp_sum(nn) * (1.f / 32.f);
  float dv = nn - mu;
  float var = warp_sum(dv * dv) * (1.f / 32.f);
  float ln = dv / (sqrtf(var) + 1e-5f) * g[lane] + b[lane];
  float s = fmaxf(ln, 0.f) / (nn + 1e-8f);
  for (int m = 0; m < M; ++m) out[base + m] = f[base + m] * s;
}

__global__ void finish_kernel(const float* acc, const float* deg, const float* si,
                              float* out, int N, int C, int M) {
  long i = (long)blockIdx.x * blockDim.x + threadIdx.x;
  long tot = (long)N * C * M;
  if (i >= tot) return;
  int n = (int)(i / (C * M));
  float d = fmaxf(deg[n], 1.f);
  out[i] = acc[i] / d + (si ? si[i] : 0.f);
}

// ---------------------------------------------------------------------------
// Host-side orchestration
// ---------------------------------------------------------------------------
struct RadP { const float *W1, *W2, *b1, *b2; };
struct LayerP {
  RadP k[4]; int nk;
  const float *proj0, *proj1, *q0, *q1;
  RadP v[4]; int nv;
  const float *nb0, *ng0, *nb1, *ng1;
};
struct BlockP {
  RadP ce[4]; int nce;
  const float *si0, *si1;
  LayerP lay[4]; int L; bool atm;
};
struct Graph { const float* rel; const float* w; const int* src; const int* dst; int E; };
struct Scratch {
  float *q0, *q1, *logits, *ex, *attn, *den, *agg0, *agg1;
  float *hA0, *hB0, *hA1, *hB1, *acc0, *acc1, *deg, *sib0, *sib1;
  unsigned* mx;
  uint32_t* pkW[4]; float* pkR[4];   // per-pair packed radial weights
  uint32_t* pkG;                     // packed B for the generic GEMM
};

struct Cursor { void* const* din; int i; const float* f() { return (const float*)din[i++]; } };
static RadP rad(Cursor& c) { RadP r; r.W1 = c.f(); r.W2 = c.f(); r.b1 = c.f(); r.b2 = c.f(); return r; }

// Leaves assumed in JAX sorted-key pytree order (dict keys sorted, lists in order).
static BlockP walk_block(Cursor& c, bool atm, int L) {
  BlockP B{}; B.L = L; B.atm = atm;
  if (atm) { B.ce[0] = rad(c); B.ce[1] = rad(c); B.ce[2] = rad(c); B.ce[3] = rad(c); B.nce = 4;
             B.si0 = c.f(); B.si1 = c.f(); }
  else     { B.ce[0] = rad(c); B.ce[1] = rad(c); B.nce = 2; B.si0 = c.f(); B.si1 = nullptr; }
  for (int l = 0; l < L; ++l) { B.lay[l].nb0 = c.f(); B.lay[l].ng0 = c.f();
                                B.lay[l].nb1 = c.f(); B.lay[l].ng1 = c.f(); }
  for (int l = 0; l < L; ++l) {
    LayerP& P = B.lay[l];
    bool d1 = atm || l > 0;
    if (d1) { P.k[0] = rad(c); P.k[1] = rad(c); P.k[2] = rad(c); P.k[3] = rad(c); P.nk = 4; }
    else    { P.k[0] = rad(c); P.nk = 1; }
    P.proj0 = c.f(); P.proj1 = c.f();
    P.q0 = c.f(); P.q1 = d1 ? c.f() : nullptr;
    if (d1) { P.v[0] = rad(c); P.v[1] = rad(c); P.v[2] = rad(c); P.v[3] = rad(c); P.nv = 4; }
    else    { P.v[0] = rad(c); P.v[1] = rad(c); P.nv = 2; }
  }
  return B;
}

static int gz(long n) { return (int)((n + 255) / 256); }
static int edge_grid(int E) { int t = (E + 15) / 16; return (t + 3) / 4; }

static void gemm(hipStream_t st, uint32_t* pk, int M, int N, int K,
                 const float* A, long lar, long lac,
                 const float* B, long lbr, long lbc,
                 const float* bias, int act,
                 float* C, long lcr, long lcc,
                 const float* A2 = nullptr, long la2r = 0, long la2c = 0, int K1 = 0,
                 const float* Cadd = nullptr, long lcar = 0, long lcac = 0) {
  int tilesN = (N + 15) / 16, kch = (K + 31) / 32;
  int pw = tilesN * kch;
  packw_kernel<<<(pw + 3) / 4, 128, 0, st>>>(B, lbr, lbc, K, N, pk);
  GemmArgs g;
  g.A = A; g.lar = lar; g.lac = lac;
  g.A2 = A2; g.la2r = la2r; g.la2c = la2c; g.K1 = A2 ? K1 : K;
  g.Bpk = pk;
  g.bias = bias; g.Cadd = Cadd; g.lcar = lcar; g.lcac = lcac;
  g.C = C; g.lcr = lcr; g.lcc = lcc;
  g.M = M; g.N = N; g.K = K; g.act = act;
  int waves = ((M + 15) / 16) * tilesN;
  gemm_kernel<<<(waves + 3) / 4, 128, 0, st>>>(g);
}

static void fillN(hipStream_t st, float* p, float v, long n) {
  fill_kernel<<<gz(n), 256, 0, st>>>(p, v, n);
}

static EdgePair mkpair(hipStream_t st, const RadP& r, int di, int dof, int cin, int cout,
                       uint32_t* wbuf, float* rbuf) {
  int nJ = (di == 1 && dof == 1) ? 3 : 1;
  int nout = nJ * cout * cin;
  int tiles = nout >> 4;
  packw_kernel<<<(tiles + 3) / 4, 128, 0, st>>>(r.W2, (long)nout, 1, 32, nout, wbuf);
  packr_kernel<<<2, 256, 0, st>>>(r.W1, r.b1, rbuf);
  EdgePair p; p.b2 = r.b2; p.W2pk = wbuf; p.R1pk = rbuf;
  p.cin = cin; p.cout = cout;
  p.cinLog = (cin == 32) ? 5 : 0;
  p.coutLog = __builtin_ctz(cout);
  p.nout = nout; p.di = di; p.dof = dof;
  return p;
}

static void run_block(hipStream_t st, const BlockP& B, int N, const Graph& G,
                      const float* h0in, const float* h1in, int c1in,
                      Scratch& S, float* out0, float* out1) {
  const float* h0c = h0in; const float* h1c = h1in; int c1 = c1in;
  for (int l = 0; l < B.L; ++l) {
    const LayerP& P = B.lay[l];
    bool d1 = (h1c != nullptr);
    // q = h @ Wq
    gemm(st, S.pkG, N, 16, 32, h0c, 32, 1, P.q0, 16, 1, nullptr, 0, S.q0, 16, 1);
    if (d1)
      for (int m = 0; m < 3; ++m)
        gemm(st, S.pkG, N, 16, c1, h1c + m, (long)c1 * 3, 3, P.q1, 16, 1, nullptr, 0, S.q1 + m, 48, 3);
    // attention logits
    EdgeArgs ea{}; ea.rel = G.rel; ea.w = G.w; ea.src = G.src; ea.dst = G.dst; ea.E = G.E;
    ea.h0 = h0c; ea.h1 = h1c; ea.c1 = d1 ? c1 : 0;
    int np = 0;
    ea.pr[np] = mkpair(st, P.k[0], 0, 0, 32, 16, S.pkW[np], S.pkR[np]); np++;
    if (d1) {
      ea.pr[np] = mkpair(st, P.k[1], 0, 1, 32, 16, S.pkW[np], S.pkR[np]); np++;
      ea.pr[np] = mkpair(st, P.k[2], 1, 0, c1, 16, S.pkW[np], S.pkR[np]); np++;
      ea.pr[np] = mkpair(st, P.k[3], 1, 1, c1, 16, S.pkW[np], S.pkR[np]); np++;
    }
    ea.np = np; ea.q0 = S.q0; ea.q1 = S.q1; ea.shared1 = d1 ? 1 : 0;
    ea.isdk = d1 ? (1.f / sqrtf(32.f)) : (1.f / sqrtf(8.f));
    ea.logits = S.logits;
    int eb = edge_grid(G.E);
    edge_kernel<0, 16, 16><<<eb, 128, 0, st>>>(ea);
    // segment softmax over dst
    long EH = (long)G.E * 2;
    fillN(st, (float*)S.mx, 0.f, (long)N * 2);          // encoded-uint 0 == -inf
    segmax_kernel<<<gz(EH), 256, 0, st>>>(S.logits, G.dst, S.mx, EH);
    fillN(st, S.den, 0.f, (long)N * 2);
    segexp_kernel<<<gz(EH), 256, 0, st>>>(S.logits, G.dst, S.mx, S.ex, S.den, EH);
    attnnorm_kernel<<<gz(EH), 256, 0, st>>>(S.ex, G.dst, S.den, S.attn, EH);
    // attention-weighted value scatter
    fillN(st, S.agg0, 0.f, (long)N * 16);
    fillN(st, S.agg1, 0.f, (long)N * 48);
    EdgeArgs ev = ea;
    int nv = 0;
    ev.pr[nv] = mkpair(st, P.v[0], 0, 0, 32, 16, S.pkW[nv], S.pkR[nv]); nv++;
    ev.pr[nv] = mkpair(st, P.v[1], 0, 1, 32, 16, S.pkW[nv], S.pkR[nv]); nv++;
    if (d1) {
      ev.pr[nv] = mkpair(st, P.v[2], 1, 0, c1, 16, S.pkW[nv], S.pkR[nv]); nv++;
      ev.pr[nv] = mkpair(st, P.v[3], 1, 1, c1, 16, S.pkW[nv], S.pkR[nv]); nv++;
    }
    ev.np = nv; ev.attn = S.attn; ev.agg0 = S.agg0; ev.agg1 = S.agg1;
    edge_kernel<1, 16, 16><<<eb, 128, 0, st>>>(ev);
    // projection over concat(agg, h)
    float* o0 = (l & 1) ? S.hB0 : S.hA0;
    float* o1 = (l & 1) ? S.hB1 : S.hA1;
    gemm(st, S.pkG, N, 32, 48, S.agg0, 16, 1, P.proj0, 32, 1, nullptr, 0, o0, 32, 1, h0c, 32, 1, 16);
    for (int m = 0; m < 3; ++m) {
      if (d1) gemm(st, S.pkG, N, 32, 16 + c1, S.agg1 + m, 48, 3, P.proj1, 32, 1, nullptr, 0,
                   o1 + m, 96, 3, h1c + m, (long)c1 * 3, 3, 16);
      else    gemm(st, S.pkG, N, 32, 16, S.agg1 + m, 48, 3, P.proj1, 32, 1, nullptr, 0, o1 + m, 96, 3);
    }
    int gb = (N + 3) / 4;
    gnorm_kernel<<<gb, 128, 0, st>>>(o0, o0, P.ng0, P.nb0, N, 1);
    gnorm_kernel<<<gb, 128, 0, st>>>(o1, o1, P.ng1, P.nb1, N, 3);
    h0c = o0; h1c = o1; c1 = 32;
  }
  // final graph conv
  int C0 = B.atm ? 64 : 32;
  fillN(st, S.acc0, 0.f, (long)N * C0);
  fillN(st, S.deg, 0.f, N);
  if (B.atm) fillN(st, S.acc1, 0.f, (long)N * 3);
  EdgeArgs ec{}; ec.rel = G.rel; ec.w = G.w; ec.src = G.src; ec.dst = G.dst; ec.E = G.E;
  ec.h0 = h0c; ec.h1 = h1c; ec.c1 = 32;
  if (B.atm) {
    ec.pr[0] = mkpair(st, B.ce[0], 0, 0, 32, 64, S.pkW[0], S.pkR[0]);
    ec.pr[1] = mkpair(st, B.ce[1], 0, 1, 32, 1,  S.pkW[1], S.pkR[1]);
    ec.pr[2] = mkpair(st, B.ce[2], 1, 0, 32, 64, S.pkW[2], S.pkR[2]);
    ec.pr[3] = mkpair(st, B.ce[3], 1, 1, 32, 1,  S.pkW[3], S.pkR[3]);
    ec.np = 4;
  } else {
    ec.pr[0] = mkpair(st, B.ce[0], 0, 0, 32, 32, S.pkW[0], S.pkR[0]);
    ec.pr[1] = mkpair(st, B.ce[1], 1, 0, 32, 32, S.pkW[1], S.pkR[1]);
    ec.np = 2;
  }
  ec.acc0 = S.acc0; ec.acc1 = S.acc1; ec.deg = S.deg;
  int eb = edge_grid(G.E);
  if (B.atm) edge_kernel<2, 64, 1><<<eb, 128, 0, st>>>(ec);
  else       edge_kernel<2, 32, 0><<<eb, 128, 0, st>>>(ec);
  // self-interaction + degree normalization
  gemm(st, S.pkG, N, C0, 32, h0c, 32, 1, B.si0, C0, 1, nullptr, 0, S.sib0, C0, 1);
  if (B.atm)
    for (int m = 0; m < 3; ++m)
      gemm(st, S.pkG, N, 1, 32, h1c + m, 96, 3, B.si1, 1, 1, nullptr, 0, S.sib1 + m, 3, 1);
  finish_kernel<<<gz((long)N * C0), 256, 0, st>>>(S.acc0, S.deg, S.sib0, out0, N, C0, 1);
  if (B.atm)
    finish_kernel<<<gz((long)N * 3), 256, 0, st>>>(S.acc1, S.deg, S.sib1, out1, N, 1, 3);
}

extern "C" void kernel_launch(void* const* d_in, const int* in_sizes, int n_in,
                              void* d_out, int out_size, void* d_ws, size_t ws_size,
                              hipStream_t stream) {
  (void)n_in; (void)out_size; (void)ws_size;
  const float* f_bnd = (const float*)d_in[0];
  const float* f_res = (const float*)d_in[1];
  const float* x_atm = (const float*)d_in[2];
  Graph GB{(const float*)d_in[3],  (const float*)d_in[4],  (const int*)d_in[5],  (const int*)d_in[6],  in_sizes[5]};
  Graph GR{(const float*)d_in[7],  (const float*)d_in[8],  (const int*)d_in[9],  (const int*)d_in[10], in_sizes[9]};
  Graph GA{(const float*)d_in[11], (const float*)d_in[12], (const int*)d_in[13], (const int*)d_in[14], in_sizes[13]};
  const float* r2a = (const float*)d_in[15];
  const float* lig = (const float*)d_in[16];
  const int NA = in_sizes[0] / 95, NR = in_sizes[1] / 29;
  const int Lg = in_sizes[16] / NA;
  const int EA = GA.E;

  Cursor cu{d_in, 17};
  const float* c1W = cu.f(); const float* c1b = cu.f();
  const float* c2W = cu.f(); const float* c2b = cu.f();
  BlockP Batm = walk_block(cu, true, 4);
  BlockP Bbnd = walk_block(cu, false, 2);
  BlockP Bres = walk_block(cu, false, 4);
  const float* l1aW = cu.f(); const float* l1ab = cu.f();
  const float* l1bW = cu.f(); const float* l1bb = cu.f();
  const float* l1rW = cu.f(); const float* l1rb = cu.f();
  const float* l2aW = cu.f(); const float* l2ab = cu.f();
  const float* l2bW = cu.f(); const float* l2bb = cu.f();
  const float* l2rW = cu.f(); const float* l2rb = cu.f();

  // bump allocator over d_ws (~70 MB needed)
  char* wsb = (char*)d_ws; size_t off = 0;
  auto alloc = [&](size_t elems) -> float* {
    float* p = (float*)(wsb + off);
    off += (elems * 4 + 255) & ~(size_t)255;
    return p;
  };
  Scratch S;
  S.q0 = alloc((size_t)NA * 16);  S.q1 = alloc((size_t)NA * 48);
  S.logits = alloc((size_t)EA * 2); S.ex = alloc((size_t)EA * 2); S.attn = alloc((size_t)EA * 2);
  S.mx = (unsigned*)alloc((size_t)NA * 2); S.den = alloc((size_t)NA * 2);
  S.agg0 = alloc((size_t)NA * 16); S.agg1 = alloc((size_t)NA * 48);
  S.hA0 = alloc((size_t)NA * 32); S.hB0 = alloc((size_t)NA * 32);
  S.hA1 = alloc((size_t)NA * 96); S.hB1 = alloc((size_t)NA * 96);
  S.acc0 = alloc((size_t)NA * 64); S.acc1 = alloc((size_t)NA * 3);
  S.deg = alloc(NA); S.sib0 = alloc((size_t)NA * 64); S.sib1 = alloc((size_t)NA * 3);
  for (int i = 0; i < 4; ++i) {
    S.pkW[i] = (uint32_t*)alloc(32768);   // max nout=2048 -> 128 tiles * 256 dwords
    S.pkR[i] = alloc(2048);
  }
  S.pkG = (uint32_t*)alloc(131072);       // max K=16000, N<=16 -> 500*256 dwords
  float* t95   = alloc((size_t)NA * 95);
  float* t64   = alloc((size_t)NA * 64);
  float* h0in  = alloc((size_t)NA * 32);
  float* hbnd  = alloc((size_t)NA * 32);
  float* hres  = alloc((size_t)NR * 32);
  float* hresA = alloc((size_t)NA * 32);
  float* l0    = alloc((size_t)NA * 32);
  float* ha0   = alloc((size_t)NA * 64);
  float* ha1   = alloc((size_t)NA * 3);
  float* cat   = alloc((size_t)NA * 10);

  // ---- bond graph ----
  gemm(stream, S.pkG, NA, 95, 95, f_bnd, 95, 1, l1bW, 95, 1, l1bb, 2, t95, 95, 1);
  gemm(stream, S.pkG, NA, 32, 95, t95, 95, 1, l2bW, 32, 1, l2bb, 0, h0in, 32, 1);
  run_block(stream, Bbnd, NA, GB, h0in, nullptr, 0, S, hbnd, nullptr);
  // ---- residue graph ----
  gemm(stream, S.pkG, NR, 29, 29, f_res, 29, 1, l1rW, 29, 1, l1rb, 2, t95, 29, 1);
  gemm(stream, S.pkG, NR, 32, 29, t95, 29, 1, l2rW, 32, 1, l2rb, 0, h0in, 32, 1);
  run_block(stream, Bres, NR, GR, h0in, nullptr, 0, S, hres, nullptr);
  gemm(stream, S.pkG, NA, 32, NR, r2a, NR, 1, hres, 32, 1, nullptr, 0, hresA, 32, 1);
  // l0 = lin2(elu(lin1(concat(h_bnd, h_resA))))
  gemm(stream, S.pkG, NA, 64, 64, hbnd, 32, 1, l1aW, 64, 1, l1ab, 2, t64, 64, 1, hresA, 32, 1, 32);
  gemm(stream, S.pkG, NA, 32, 64, t64, 64, 1, l2aW, 32, 1, l2ab, 0, l0, 32, 1);
  // ---- atom graph ----
  run_block(stream, Batm, NA, GA, l0, x_atm, 1, S, ha0, ha1);
  // readout
  gemm(stream, S.pkG, NA, 64, 64, ha0, 64, 1, c1W, 64, 1, c1b, 1, t64, 64, 1);
  gemm(stream, S.pkG, NA, 10, 64, t64, 64, 1, c2W, 10, 1, c2b, 0, cat, 10, 1);
  float* outp = (float*)d_out;
  gemm(stream, S.pkG, Lg, 10, NA, lig, NA, 1, cat, 10, 1, nullptr, 0, outp, 10, 1);
  gemm(stream, S.pkG, Lg, 3,  NA, lig, NA, 1, ha1, 3, 1, nullptr, 0, outp + (size_t)Lg * 10, 3, 1);
}
